// MoE_21792664060307
// MI455X (gfx1250) — compile-verified, matrix operand order checked by reading, not run on hardware
//
#include <hip/hip_runtime.h>
#include <hip/hip_bf16.h>
#include <math.h>

// ---------------------------------------------------------------------------
// MoE forward for MI455X (gfx1250, wave32, WMMA).
// Top-2 gather dispatch (exact vs. the dense reference, since unselected
// experts have probability 0) + always-on static expert.
//
// Pipeline:
//   0. repack:  w1/w2/w1s/w2s f32 -> bf16 in WMMA-B-fragment order (once;
//               removes all loads-strided-by-row and all f32->bf16 converts
//               from the GEMM inner loop; bf16 weights ~134MB fit in 192MB L2)
//   0b. xcast:  x f32 -> bf16 row-major (so the token gather is a raw copy
//               and can use the gfx1250 async global->LDS path / ASYNCcnt)
//   1. gate:    scores, top-2, softmax
//   2. aux:     deterministic tree-reduced variance penalty -> d_out[N*C]
//   3. lists:   per-expert token compaction via wave32 ballot prefix sums
//   4. moe_gemm: fused  y = (gelu(x @ W1) @ W2) * weight  on
//               v_wmma_f32_16x16x32_bf16, f32 accumulate, LDS-staged tiles
//   5. combine: y = y_static + contrib[slot0] + contrib[slot1]
// Deterministic: no float atomics anywhere.
// ---------------------------------------------------------------------------

typedef __attribute__((ext_vector_type(16))) __bf16 v16bf;
typedef __attribute__((ext_vector_type(8)))  __bf16 v8bf;
typedef __attribute__((ext_vector_type(4)))  __bf16 v4bf;
typedef __attribute__((ext_vector_type(8)))  float  v8f;
typedef __attribute__((ext_vector_type(4)))  int    v4i;

constexpr int   Bc = 2;
constexpr int   Tc = 2048;
constexpr int   Cc = 1024;           // n_embd
constexpr int   Hc = 4096;           // MLP hidden
constexpr int   Ec = 7;              // dynamic experts
constexpr int   Nn = Bc * Tc;        // 4096 tokens
constexpr int   MT = 64;             // token tile (rows per block)
constexpr int   KH = 64;             // H-chunk for the fused 2-stage GEMM
constexpr float MOE_LOSS_COEF = 0.01f;

#ifndef __has_builtin
#define __has_builtin(x) 0
#endif
#if __has_builtin(__builtin_amdgcn_global_load_async_to_lds_b128)
#define HAVE_ASYNC_LDS 1
#else
#define HAVE_ASYNC_LDS 0
#endif

#if HAVE_ASYNC_LDS
typedef __attribute__((address_space(1))) v4i* gv4i_p;   // global int4*
typedef __attribute__((address_space(3))) v4i* sv4i_p;   // LDS int4*
#endif

__device__ __forceinline__ v8f wmma_bf16(v16bf a, v16bf b, v8f c) {
  // (neg_a, A, neg_b, B, c_mod, C, reuse_a, reuse_b)
  return __builtin_amdgcn_wmma_f32_16x16x32_bf16(false, a, false, b,
                                                 (short)0, c, false, false);
}

__device__ __forceinline__ v8f vzero8() {
  v8f z;
#pragma unroll
  for (int i = 0; i < 8; ++i) z[i] = 0.0f;
  return z;
}

__device__ __forceinline__ float gelu_exact(float v) {
  // 0.5 * v * (1 + erf(v / sqrt(2)))  (approximate=False in the reference)
  return 0.5f * v * (1.0f + erff(v * 0.70710678118654752f));
}

// ---------------------------------------------------------------------------
// Kernel 0: repack a [K x N] f32 row-major weight matrix (E matrices) into
// bf16 WMMA-B-fragment order.  Tile (kt, nt) covers K rows [kt*32, +32) and
// cols [nt*16, +16).  Packed tile t = nt*(K/32) + kt occupies 512 halves;
// lane l owns 16 contiguous halves at t*512 + l*16:
//   value(l, j) = W[kt*32 + koff(l>>4, j)][nt*16 + (l&15)],
//   koff(h, j) = h*8 + j (j<8),  16 + h*8 + (j-8) (j>=8)
// so the GEMM reads one aligned 32B v16bf per fragment (2x global_load_b128,
// 32 lanes x 32B = 1KB contiguous per tile).
// ---------------------------------------------------------------------------
__global__ __launch_bounds__(256) void repack_kernel(
    const float* __restrict__ src, __bf16* __restrict__ dst,
    int K, int N, int matStride /* = K*N */) {
  const size_t gid = (size_t)blockIdx.x * 256 + threadIdx.x;
  const int    lane = (int)(gid & 31);
  const size_t tileId = gid >> 5;
  const int    tilesPerMat = (K / 32) * (N / 16);
  const int    e  = (int)(tileId / tilesPerMat);
  const int    t  = (int)(tileId % tilesPerMat);
  const int    kt = t % (K / 32);
  const int    nt = t / (K / 32);
  const int    ml = lane & 15, lh = lane >> 4;
  const int    k0 = kt * 32, c0 = nt * 16 + ml;

  const float* S = src + (size_t)e * matStride;
  v16bf o;
#pragma unroll
  for (int j = 0; j < 8; ++j) {
    o[j]     = (__bf16)S[(size_t)(k0 + lh * 8 + j)      * N + c0];
    o[j + 8] = (__bf16)S[(size_t)(k0 + 16 + lh * 8 + j) * N + c0];
  }
  *(v16bf*)(dst + (size_t)e * matStride + (size_t)t * 512 + lane * 16) = o;
}

// Kernel 0b: x f32 -> bf16, row-major (4 elements / thread).
__global__ __launch_bounds__(256) void xcast_kernel(
    const float* __restrict__ x, __bf16* __restrict__ xb) {
  const size_t i = ((size_t)blockIdx.x * 256 + threadIdx.x) * 4;
  const float4 v = *(const float4*)(x + i);
  v4bf o;
  o[0] = (__bf16)v.x; o[1] = (__bf16)v.y; o[2] = (__bf16)v.z; o[3] = (__bf16)v.w;
  *(v4bf*)(xb + i) = o;
}

// ---------------------------------------------------------------------------
// Kernel 1: gating.  scores = x @ gate_w  (N x 7), top-2, softmax over the 2.
// ---------------------------------------------------------------------------
__global__ __launch_bounds__(256) void gate_kernel(
    const float* __restrict__ x, const float* __restrict__ gw,
    int* __restrict__ tidx, float* __restrict__ tw) {
  const int n = blockIdx.x * 256 + threadIdx.x;
  if (n >= Nn) return;
  const float* xr = x + (size_t)n * Cc;

  float s[Ec];
#pragma unroll
  for (int e = 0; e < Ec; ++e) s[e] = 0.0f;

  for (int c = 0; c < Cc; c += 4) {
    const float4 xv = *(const float4*)(xr + c);
#pragma unroll
    for (int e = 0; e < Ec; ++e) {
      s[e] = fmaf(xv.x, gw[(c + 0) * Ec + e], s[e]);
      s[e] = fmaf(xv.y, gw[(c + 1) * Ec + e], s[e]);
      s[e] = fmaf(xv.z, gw[(c + 2) * Ec + e], s[e]);
      s[e] = fmaf(xv.w, gw[(c + 3) * Ec + e], s[e]);
    }
  }
  int i0 = 0;                       // strict > keeps lowest index on ties
#pragma unroll
  for (int e = 1; e < Ec; ++e) if (s[e] > s[i0]) i0 = e;
  int i1 = (i0 == 0) ? 1 : 0;
#pragma unroll
  for (int e = 0; e < Ec; ++e) if (e != i0 && s[e] > s[i1]) i1 = e;

  const float m   = fmaxf(s[i0], s[i1]);
  const float e0  = expf(s[i0] - m);
  const float e1  = expf(s[i1] - m);
  const float inv = 1.0f / (e0 + e1);

  tidx[n * 2 + 0] = i0;  tidx[n * 2 + 1] = i1;
  tw[n * 2 + 0] = e0 * inv;  tw[n * 2 + 1] = e1 * inv;
}

// ---------------------------------------------------------------------------
// Kernel 2: auxiliary variance-penalty loss (deterministic tree reduction).
// ---------------------------------------------------------------------------
__global__ __launch_bounds__(256) void aux_kernel(
    const int* __restrict__ tidx, const float* __restrict__ tw,
    float* __restrict__ yout) {
  __shared__ float red[256][8];
  float u[Ec];
#pragma unroll
  for (int e = 0; e < Ec; ++e) u[e] = 0.0f;

  for (int t = threadIdx.x; t < Nn; t += 256) {
#pragma unroll
    for (int k = 0; k < 2; ++k) {
      const int   id = tidx[t * 2 + k];
      const float w  = tw[t * 2 + k];
#pragma unroll
      for (int e = 0; e < Ec; ++e) u[e] += (id == e) ? w : 0.0f;
    }
  }
#pragma unroll
  for (int e = 0; e < Ec; ++e) red[threadIdx.x][e] = u[e];
  __syncthreads();
  for (int off = 128; off > 0; off >>= 1) {
    if (threadIdx.x < (unsigned)off)
#pragma unroll
      for (int e = 0; e < Ec; ++e)
        red[threadIdx.x][e] += red[threadIdx.x + off][e];
    __syncthreads();
  }
  if (threadIdx.x == 0) {
    float us[Ec], mean = 0.0f;
#pragma unroll
    for (int e = 0; e < Ec; ++e) { us[e] = red[0][e] / (float)Nn; mean += us[e]; }
    mean /= (float)Ec;
    float var = 0.0f;
#pragma unroll
    for (int e = 0; e < Ec; ++e) var += (us[e] - mean) * (us[e] - mean);
    var /= (float)Ec;
    yout[(size_t)Nn * Cc] = MOE_LOSS_COEF * var;   // aux appended after y
  }
}

// ---------------------------------------------------------------------------
// Kernel 3: deterministic per-expert token-list compaction.
// One wave32 per expert; ballot + popc prefix in token order (no atomics).
// ---------------------------------------------------------------------------
__global__ __launch_bounds__(256) void build_lists_kernel(
    const int* __restrict__ tidx, const float* __restrict__ tw,
    int* __restrict__ tok_list, float* __restrict__ wt_list,
    int* __restrict__ ks_list, int* __restrict__ counts) {
  const int wave = threadIdx.x >> 5;
  const int lane = threadIdx.x & 31;
  if (wave >= Ec) return;

  int cnt = 0;
  for (int base = 0; base < Nn; base += 32) {
    const int tok = base + lane;
    const int j0 = tidx[tok * 2 + 0];
    const int j1 = tidx[tok * 2 + 1];
    const bool p0 = (j0 == wave);
    const bool p1 = (j1 == wave);
    const bool p  = p0 || p1;
    const unsigned mask = (unsigned)__ballot(p);   // wave32: bits [31:0]
    if (p) {
      const int slot = cnt + __popc(mask & ((1u << lane) - 1u));
      tok_list[wave * Nn + slot] = tok;
      wt_list [wave * Nn + slot] = p0 ? tw[tok * 2 + 0] : tw[tok * 2 + 1];
      ks_list [wave * Nn + slot] = p0 ? 0 : 1;     // which top-k slot
    }
    cnt += __popc(mask);
  }
  if (lane == 0) counts[wave] = cnt;
}

// ---------------------------------------------------------------------------
// Kernel 4: fused expert MLP,  y_tile = (gelu(x_tile @ W1) @ W2) * weight.
// Grid: (N/MT token tiles) x (7 dynamic experts + 1 static).
// Block = 256 threads = 8 waves; 64x1024 f32 output tile in VGPRs
// (8 waves x 128 cols -> 4x8 accum tiles = 256 VGPRs/lane).
// Hot loop per WMMA: 2x ds_load_b128 (A) + 2x global_load_b128 (B, packed
// bf16) + v_wmma_f32_16x16x32_bf16.  No converts, no strided scalar loads.
// ---------------------------------------------------------------------------
__global__ __launch_bounds__(256) void moe_gemm_kernel(
    const __bf16* __restrict__ xbf,
    const __bf16* __restrict__ w1p,  const __bf16* __restrict__ w2p,
    const __bf16* __restrict__ w1sp, const __bf16* __restrict__ w2sp,
    const int* __restrict__ tok_list, const float* __restrict__ wt_list,
    const int* __restrict__ ks_list,  const int* __restrict__ counts,
    float* __restrict__ ycontrib, float* __restrict__ yout) {
  // LDS: 64x(1024+8) bf16 x-tile (129KB) + 64x(64+8) bf16 h-tile (9KB)
  // of the 320KB WGP LDS; all fragment rows stay 16B-aligned.
  __shared__ __bf16 xs[MT][Cc + 8];
  __shared__ __bf16 hs[MT][KH + 8];
  __shared__ int    stok[MT];
  __shared__ float  swt[MT];
  __shared__ int    sks[MT];

  const int  e        = blockIdx.y;        // 0..6 dynamic, 7 = static
  const bool isStatic = (e == Ec);
  const int  m        = blockIdx.x;
  const int  cnt      = isStatic ? Nn : counts[e];
  if (m * MT >= cnt) return;               // block-uniform exit
  const int rowcnt = min(MT, cnt - m * MT);

  const int tid  = threadIdx.x;
  const int lane = tid & 31;
  const int wave = tid >> 5;
  const int ml   = lane & 15;              // position within 16 (M or N)
  const int lh   = lane >> 4;              // lane-half selects K split

  if (tid < MT) {
    const int r = tid;
    int tok = -1, ks = 0; float wg = 0.0f;
    if (r < rowcnt) {
      if (isStatic) { tok = m * MT + r; wg = 1.0f; }
      else {
        const int s = m * MT + r;
        tok = tok_list[e * Nn + s];
        wg  = wt_list [e * Nn + s];
        ks  = ks_list [e * Nn + s];
      }
    }
    stok[r] = tok; swt[r] = wg; sks[r] = ks;
  }
  __syncthreads();

  // Gather bf16 token rows into LDS: raw byte move, 4 threads per row.
  {
    const int r   = tid >> 2;
    const int seg = tid & 3;                       // 256 halves per segment
    const int tok = stok[r];
    if (tok >= 0) {
      const __bf16* src = xbf + (size_t)tok * Cc + seg * 256;
#if HAVE_ASYNC_LDS
      // gfx1250 async global->LDS DMA path (ASYNCcnt), 16B per op.
      __bf16* srcnc = const_cast<__bf16*>(src);
#pragma unroll
      for (int j = 0; j < 32; ++j)
        __builtin_amdgcn_global_load_async_to_lds_b128(
            (gv4i_p)(void*)(srcnc + j * 8),
            (sv4i_p)(void*)&xs[r][seg * 256 + j * 8],
            0, 0);
#else
      const unsigned long long* s8 = (const unsigned long long*)src;
      unsigned long long* d8 = (unsigned long long*)&xs[r][seg * 256];
#pragma unroll 8
      for (int j = 0; j < 64; ++j) d8[j] = s8[j];
#endif
    } else {
      unsigned long long* d8 = (unsigned long long*)&xs[r][seg * 256];
#pragma unroll 8
      for (int j = 0; j < 64; ++j) d8[j] = 0ull;
    }
  }
#if HAVE_ASYNC_LDS
#if __has_builtin(__builtin_amdgcn_s_wait_asynccnt)
  __builtin_amdgcn_s_wait_asynccnt(0);
#else
  asm volatile("s_wait_asynccnt 0" ::: "memory");
#endif
#endif
  __syncthreads();

  // Packed bf16 weight bases (fragment order, see repack_kernel).
  const __bf16* W1 = isStatic ? w1sp : (w1p + (size_t)e * Cc * Hc);
  const __bf16* W2 = isStatic ? w2sp : (w2p + (size_t)e * Hc * Cc);

  v8f acc[4][8];
#pragma unroll
  for (int rt = 0; rt < 4; ++rt)
#pragma unroll
    for (int ct = 0; ct < 8; ++ct) acc[rt][ct] = vzero8();

  const int cbase = wave * 128;            // this wave's 128 output columns

  for (int hb = 0; hb < Hc; hb += KH) {
    // ---- stage 1: hs = gelu(xs @ W1[:, hb:hb+KH]); 16 tiles, 2 per wave --
#pragma unroll
    for (int t = 0; t < 2; ++t) {
      const int tile = wave * 2 + t;       // 0..15
      const int tr = tile & 3;             // token-row tile
      const int tc = tile >> 2;            // h-column tile
      // packed column group for this tile, k-major
      const __bf16* wtile =
          W1 + ((size_t)(hb / 16 + tc) * (Cc / 32)) * 512 + lane * 16;
      v8f h = vzero8();
      for (int k = 0; k < Cc; k += 32) {
        const v8bf alo = *(const v8bf*)&xs[tr * 16 + ml][k + lh * 8];
        const v8bf ahi = *(const v8bf*)&xs[tr * 16 + ml][k + 16 + lh * 8];
        v16bf a;
#pragma unroll
        for (int j = 0; j < 8; ++j) { a[j] = alo[j]; a[j + 8] = ahi[j]; }
        const v16bf b = *(const v16bf*)(wtile + (size_t)(k >> 5) * 512);
        if (k + 32 < Cc)
          __builtin_prefetch(wtile + (size_t)((k >> 5) + 1) * 512, 0, 1);
        h = wmma_bf16(a, b, h);
      }
      // D tile layout: VGPR i -> row tr*16 + i + lh*8, col tc*16 + ml
#pragma unroll
      for (int i = 0; i < 8; ++i)
        hs[tr * 16 + i + lh * 8][tc * 16 + ml] = (__bf16)gelu_exact(h[i]);
    }
    __syncthreads();

    // ---- stage 2: acc += hs @ W2[hb:hb+KH, cbase:cbase+128] --------------
#pragma unroll
    for (int k2 = 0; k2 < KH; k2 += 32) {
      const size_t kt2 = (size_t)((hb + k2) >> 5);
#pragma unroll
      for (int rt = 0; rt < 4; ++rt) {
        const v8bf alo = *(const v8bf*)&hs[rt * 16 + ml][k2 + lh * 8];
        const v8bf ahi = *(const v8bf*)&hs[rt * 16 + ml][k2 + 16 + lh * 8];
        v16bf a;
#pragma unroll
        for (int j = 0; j < 8; ++j) { a[j] = alo[j]; a[j + 8] = ahi[j]; }
#pragma unroll
        for (int ct = 0; ct < 8; ++ct) {
          const v16bf b = *(const v16bf*)(
              W2 + ((size_t)(wave * 8 + ct) * (Hc / 32) + kt2) * 512 +
              lane * 16);
          acc[rt][ct] = wmma_bf16(a, b, acc[rt][ct]);
        }
      }
    }
    __syncthreads();
  }

  // ---- epilogue: weight rows and scatter ---------------------------------
#pragma unroll
  for (int rt = 0; rt < 4; ++rt) {
#pragma unroll
    for (int i = 0; i < 8; ++i) {
      const int r = rt * 16 + i + lh * 8;
      if (r < rowcnt) {
        const int   tok = stok[r];
        const float wg  = swt[r];
        if (isStatic) {
          float* dst = yout + (size_t)tok * Cc;
#pragma unroll
          for (int ct = 0; ct < 8; ++ct)
            dst[cbase + ct * 16 + ml] = acc[rt][ct][i];          // weight == 1
        } else {
          float* dst = ycontrib + ((size_t)sks[r] * Nn + tok) * Cc;
#pragma unroll
          for (int ct = 0; ct < 8; ++ct)
            dst[cbase + ct * 16 + ml] = acc[rt][ct][i] * wg;
        }
      }
    }
  }
}

// ---------------------------------------------------------------------------
// Kernel 5: y = y_static + contrib[kslot0] + contrib[kslot1]
// (each token appears exactly once per kslot plane -> both fully written)
// ---------------------------------------------------------------------------
__global__ __launch_bounds__(256) void combine_kernel(
    const float* __restrict__ yc, float* __restrict__ yout) {
  const size_t i = (size_t)blockIdx.x * 256 + threadIdx.x;
  yout[i] += yc[i] + yc[(size_t)Nn * Cc + i];
}

// ---------------------------------------------------------------------------
extern "C" void kernel_launch(void* const* d_in, const int* in_sizes, int n_in,
                              void* d_out, int out_size, void* d_ws, size_t ws_size,
                              hipStream_t stream) {
  const float* x   = (const float*)d_in[0];
  const float* gw  = (const float*)d_in[1];
  const float* w1  = (const float*)d_in[2];
  const float* w2  = (const float*)d_in[3];
  const float* w1s = (const float*)d_in[4];
  const float* w2s = (const float*)d_in[5];
  float* yout = (float*)d_out;           // [N*C] y, then [1] aux

  char* p = (char*)d_ws;
  auto alloc = [&](size_t bytes) -> char* {
    char* r = p; p += (bytes + 255) & ~(size_t)255; return r;
  };
  int*    tidx = (int*)   alloc(sizeof(int)    * Nn * 2);
  float*  tw   = (float*) alloc(sizeof(float)  * Nn * 2);
  int*    tokl = (int*)   alloc(sizeof(int)    * Ec * Nn);
  float*  wtl  = (float*) alloc(sizeof(float)  * Ec * Nn);
  int*    ksl  = (int*)   alloc(sizeof(int)    * Ec * Nn);
  int*    cnts = (int*)   alloc(sizeof(int)    * 8);
  float*  ycon = (float*) alloc(sizeof(float)  * 2ull * Nn * Cc);   // 32 MB
  __bf16* xbf  = (__bf16*)alloc(sizeof(__bf16) * (size_t)Nn * Cc);  //  8 MB
  __bf16* w1pk = (__bf16*)alloc(sizeof(__bf16) * (size_t)Ec * Cc * Hc); // 56 MB
  __bf16* w2pk = (__bf16*)alloc(sizeof(__bf16) * (size_t)Ec * Hc * Cc); // 56 MB
  __bf16* w1sp = (__bf16*)alloc(sizeof(__bf16) * (size_t)Cc * Hc);  //  8 MB
  __bf16* w2sp = (__bf16*)alloc(sizeof(__bf16) * (size_t)Hc * Cc);  //  8 MB

  const int tilesPerMat = (Cc / 32) * (Hc / 16);   // == (Hc/32)*(Cc/16) = 8192
  repack_kernel<<<Ec * tilesPerMat * 32 / 256, 256, 0, stream>>>(
      w1, w1pk, Cc, Hc, Cc * Hc);
  repack_kernel<<<Ec * tilesPerMat * 32 / 256, 256, 0, stream>>>(
      w2, w2pk, Hc, Cc, Hc * Cc);
  repack_kernel<<<tilesPerMat * 32 / 256, 256, 0, stream>>>(
      w1s, w1sp, Cc, Hc, Cc * Hc);
  repack_kernel<<<tilesPerMat * 32 / 256, 256, 0, stream>>>(
      w2s, w2sp, Hc, Cc, Hc * Cc);
  xcast_kernel<<<(Nn * Cc / 4) / 256, 256, 0, stream>>>(x, xbf);

  gate_kernel<<<Nn / 256, 256, 0, stream>>>(x, gw, tidx, tw);
  aux_kernel<<<1, 256, 0, stream>>>(tidx, tw, yout);
  build_lists_kernel<<<1, 256, 0, stream>>>(tidx, tw, tokl, wtl, ksl, cnts);
  moe_gemm_kernel<<<dim3(Nn / MT, Ec + 1), 256, 0, stream>>>(
      xbf, w1pk, w2pk, w1sp, w2sp, tokl, wtl, ksl, cnts, ycon, yout);
  combine_kernel<<<(Nn * Cc) / 256, 256, 0, stream>>>(ycon, yout);
}
